// VectorQuantizer_71365176590879
// MI455X (gfx1250) — matrix operand order repaired and checked by reading, not compiled, not run on hardware
//
#include <hip/hip_runtime.h>
#include <hip/hip_bf16.h>
#include <math.h>

// ---------------- types for WMMA operands ----------------
typedef _Float16 v16h __attribute__((ext_vector_type(16)));
typedef _Float16 v8h  __attribute__((ext_vector_type(8)));
typedef float    v8f  __attribute__((ext_vector_type(8)));
typedef int      v4i  __attribute__((vector_size(4 * sizeof(int))));

// Problem constants
#define NPOS   65536        // 16*64*64 spatial positions
#define CCH    256          // channels / embedding dim
#define KCODE  1024         // codebook entries
#define HWIMG  4096         // 64*64
#define LOSS_POS 16777216
#define PERP_POS 16777217
#define IDX_BASE 16777218

// Block geometry: 128 threads = 4 waves; each wave owns M=32 positions
// (two 16x16x32 accumulators share every B fetch -> 2x arithmetic intensity
// on the B stream, which is the bandwidth bottleneck on MI455X).
#define TPB      128
#define POSB     128        // positions per block
#define NBLK     (NPOS / POSB)

// LDS A region: 128 rows * (256+8 pad) halves -> 528B rows, 16B aligned.
#define ROWH    264
#define A_HALFS (POSB * ROWH)          // 33792 halves = 67584 B
// LDS B region: double-buffered 16x256 f16 code tile (8KB each)
#define BTILE_H 4096                   // halves per B tile
#define B_HALFS (2 * BTILE_H)          // 16384 B
#define LDS_BYTES ((A_HALFS + B_HALFS) * 2)   // 83968 B (< 320KB WGP LDS)

// workspace layout (bytes)
#define WS_LOSS   0
#define WS_COUNTS 256        // 1024 * u32
#define WS_ENORM  4608       // 1024 * f32
#define WS_EBB    9216       // 1024*256 f16 in B-operand layout (512KB)

// CDNA5 async global->LDS path (ASYNCcnt); device pass only.
#if defined(__AMDGCN__) && __has_builtin(__builtin_amdgcn_global_load_async_to_lds_b128)
#define VQ_ASYNC 1
#endif

#ifdef VQ_ASYNC
typedef __attribute__((address_space(1))) v4i GV4;   // global int4
typedef __attribute__((address_space(3))) v4i LV4;   // LDS int4
__device__ __forceinline__ void vq_async_cp16(const _Float16* g, _Float16* l) {
    __builtin_amdgcn_global_load_async_to_lds_b128((GV4*)g, (LV4*)l, 0, 0);
}
__device__ __forceinline__ void vq_wait_async4() {
#if __has_builtin(__builtin_amdgcn_s_wait_asynccnt)
    __builtin_amdgcn_s_wait_asynccnt(4);
#else
    asm volatile("s_wait_asynccnt 0x4" ::: "memory");
#endif
}
#endif

// -----------------------------------------------------------------------------
// prep: build f16 (-2*e) codebook in WMMA B layout, ||e||^2, zero accumulators.
// B layout (V_WMMA_F32_16X16X32_F16, 32x16 KxN, wave32):
//   lane = column N (N = lane%16); this code's K order: lanes 0-15 hold K=0..15, lanes 16-31 K=16..31
// Stored per lane as 16 contiguous f16 (32B): element j <-> K = 16h + j.
// -----------------------------------------------------------------------------
__global__ void __launch_bounds__(256) vq_prep(const float* __restrict__ emb,
                                               _Float16* __restrict__ ebB,
                                               float* __restrict__ enorm2,
                                               unsigned* __restrict__ counts,
                                               float* __restrict__ lossacc) {
    int blk = blockIdx.x, t = threadIdx.x;
    if (blk < 64) {
        int g    = blk * 256 + t;             // g = (tile*8 + c)*32 + lane
        int lane = g & 31;
        int c    = (g >> 5) & 7;
        int tile = g >> 8;
        int n    = tile * 16 + (lane & 15);   // code index (column N)
        int h    = lane >> 4;
        const float* ep = emb + (size_t)n * CCH + c * 32 + 16 * h;
        _Float16* op = ebB + (size_t)g * 16;
        #pragma unroll
        for (int j = 0; j < 16; ++j) op[j] = (_Float16)(-2.0f * ep[j]);
    } else if (blk < 68) {
        int k = (blk - 64) * 256 + t;
        const float* ep = emb + (size_t)k * CCH;
        float s = 0.f;
        for (int ch = 0; ch < CCH; ++ch) { float v = ep[ch]; s += v * v; }
        enorm2[k] = s;
    } else if (blk < 72) {
        counts[(blk - 68) * 256 + t] = 0u;
    } else {
        if (t == 0) *lossacc = 0.f;
    }
}

// -----------------------------------------------------------------------------
// main
// -----------------------------------------------------------------------------
__global__ void __launch_bounds__(TPB) vq_main(const float* __restrict__ z,
                                               const float* __restrict__ emb,
                                               const _Float16* __restrict__ ebB,
                                               const float* __restrict__ enorm2,
                                               unsigned* __restrict__ counts,
                                               float* __restrict__ lossacc,
                                               float* __restrict__ out) {
    extern __shared__ _Float16 lds[];   // [A_HALFS | B double buffer]
    int t   = threadIdx.x;
    int blk = blockIdx.x;
    int n0  = blk * POSB;                // first flat position (128 | 4096 => one image)
    int b   = n0 >> 12;
    int s0  = n0 & (HWIMG - 1);
    const float* zb = z + (size_t)b * (CCH * HWIMG) + s0;
    _Float16* Bb = lds + A_HALFS;        // B staging region

    // ---- Phase 1: stage A tile (128 pos x 256 ch) to LDS as f16 -------------
    {
        int pos = t;                     // TPB == POSB
        _Float16* row = lds + (size_t)pos * ROWH;
        #pragma unroll 4
        for (int ch = 0; ch < CCH; ch += 2) {
            float f0 = zb[(size_t)ch * HWIMG + pos];
            float f1 = zb[(size_t)(ch + 1) * HWIMG + pos];
            union { _Float16 h2[2]; unsigned u; } pk;
            pk.h2[0] = (_Float16)f0; pk.h2[1] = (_Float16)f1;
            *(unsigned*)(row + ch) = pk.u;
        }
    }
#ifdef VQ_ASYNC
    {   // prologue: async copy B tile 0 into buffer 0 (64B per thread)
        const _Float16* g = ebB + (size_t)t * 32;
        _Float16* l = Bb + (size_t)t * 32;
        vq_async_cp16(g,      l);
        vq_async_cp16(g + 8,  l + 8);
        vq_async_cp16(g + 16, l + 16);
        vq_async_cp16(g + 24, l + 24);
    }
#endif
    __syncthreads();

    int lane = t & 31;
    int wave = t >> 5;
    int lm   = lane & 15;     // A: row M sub-index; B/C/D: column N sub-index
    int h    = lane >> 4;
    int m0   = wave * 32;     // this wave's 32 positions

    // ---- Preload A panel: rows [m0, m0+32) as 2 x 8 chunks (A layout) -------
    // A layout: M = lane%16 for both halves; VGPR elem j <-> K=16h+j
    // (matches B). Rows are 528B (16B aligned) -> two b128 reads.
    v16h av0[8], av1[8];
    {
        const _Float16* r0 = lds + (size_t)(m0 + lm) * ROWH;
        const _Float16* r1 = lds + (size_t)(m0 + 16 + lm) * ROWH;
        #pragma unroll
        for (int c = 0; c < 8; ++c) {
            v8h lo0 = *(const v8h*)(r0 + c * 32 + 16 * h);
            v8h hi0 = *(const v8h*)(r0 + c * 32 + 16 * h + 8);
            av0[c] = __builtin_shufflevector(lo0, hi0,
                      0, 1, 2, 3, 4, 5, 6, 7, 8, 9, 10, 11, 12, 13, 14, 15);
            v8h lo1 = *(const v8h*)(r1 + c * 32 + 16 * h);
            v8h hi1 = *(const v8h*)(r1 + c * 32 + 16 * h + 8);
            av1[c] = __builtin_shufflevector(lo1, hi1,
                      0, 1, 2, 3, 4, 5, 6, 7, 8, 9, 10, 11, 12, 13, 14, 15);
        }
    }
    __syncthreads();   // all A reads done before sIdx overlays LDS row 0

    float bestD0[8], bestD1[8];
    int   bestK0[8], bestK1[8];
    #pragma unroll
    for (int v = 0; v < 8; ++v) {
        bestD0[v] = 3.4e38f; bestK0[v] = 0;
        bestD1[v] = 3.4e38f; bestK1[v] = 0;
    }

    // ---- Phase 2: sweep 64 code tiles; D = (-2 z.e) + ||e||^2 ---------------
    for (int tn = 0; tn < 64; ++tn) {
#ifdef VQ_ASYNC
        {   // prefetch next tile into the other buffer (double buffering)
            int nt = (tn + 1) & 63;
            const _Float16* g = ebB + (size_t)nt * BTILE_H + (size_t)t * 32;
            _Float16* l = Bb + (size_t)((tn + 1) & 1) * BTILE_H + (size_t)t * 32;
            vq_async_cp16(g,      l);
            vq_async_cp16(g + 8,  l + 8);
            vq_async_cp16(g + 16, l + 16);
            vq_async_cp16(g + 24, l + 24);
        }
        vq_wait_async4();      // tile tn's 4 copies (older) are complete
        __syncthreads();       // ... in every wave -> tile tn visible block-wide
        const _Float16* Bcur = Bb + (size_t)(tn & 1) * BTILE_H;
#endif
        float en = enorm2[tn * 16 + lm];          // column N = lm for all VGPRs
        v8f acc0 = { en, en, en, en, en, en, en, en };
        v8f acc1 = acc0;
        #pragma unroll
        for (int c = 0; c < 8; ++c) {
#ifdef VQ_ASYNC
            v16h bv = *(const v16h*)(Bcur + (size_t)((c * 32 + lane) << 4));
#else
            v16h bv = *((const v16h*)(ebB + (size_t)tn * BTILE_H) + (c * 32 + lane));
#endif
            acc0 = __builtin_amdgcn_wmma_f32_16x16x32_f16(
                       false, av0[c], false, bv, (short)0, acc0, false, false);
            acc1 = __builtin_amdgcn_wmma_f32_16x16x32_f16(
                       false, av1[c], false, bv, (short)0, acc1, false, false);
        }
        int kc = tn * 16 + lm;
        #pragma unroll
        for (int v = 0; v < 8; ++v) {
            if (acc0[v] < bestD0[v]) { bestD0[v] = acc0[v]; bestK0[v] = kc; }
            if (acc1[v] < bestD1[v]) { bestD1[v] = acc1[v]; bestK1[v] = kc; }
        }
#ifdef VQ_ASYNC
        __syncthreads();       // all waves done reading buf before it is reused
#endif
    }

    // ---- Cross-lane argmin within 16-lane groups ----------------------------
    // acc0 rows: M = v + 8h at positions m0+..; acc1 rows: +16.
    #pragma unroll
    for (int v = 0; v < 8; ++v) {
        float d0 = bestD0[v]; int k0 = bestK0[v];
        float d1 = bestD1[v]; int k1 = bestK1[v];
        #pragma unroll
        for (int off = 8; off >= 1; off >>= 1) {
            float od0 = __shfl_xor(d0, off, 16);
            int   ok0 = __shfl_xor(k0, off, 16);
            if (od0 < d0 || (od0 == d0 && ok0 < k0)) { d0 = od0; k0 = ok0; }
            float od1 = __shfl_xor(d1, off, 16);
            int   ok1 = __shfl_xor(k1, off, 16);
            if (od1 < d1 || (od1 == d1 && ok1 < k1)) { d1 = od1; k1 = ok1; }
        }
        if (lm == 0) {
            ((int*)lds)[m0 + v + 8 * h]      = k0;   // overlay, bytes [0,512)
            ((int*)lds)[m0 + 16 + v + 8 * h] = k1;
        }
    }
    __syncthreads();

    // ---- Phase 3: gather z_q, straight-through output, loss, counts, idx ----
    {
        int pos = t;
        int k = ((int*)lds)[pos];
        atomicAdd(&counts[k], 1u);
        out[IDX_BASE + n0 + pos] = (float)k;
        const float* ep = emb + (size_t)k * CCH;
        float* ob = out + (size_t)b * (CCH * HWIMG) + s0;
        float ls = 0.f;
        #pragma unroll 4
        for (int ch = 0; ch < CCH; ++ch) {
            float ze = zb[(size_t)ch * HWIMG + pos];
            float q  = ep[ch];
            ob[(size_t)ch * HWIMG + pos] = ze + (q - ze);  // match reference fp
            float df = ze - q;
            ls += df * df;
        }
        #pragma unroll
        for (int off = 16; off >= 1; off >>= 1) ls += __shfl_xor(ls, off, 32);
        if (lane == 0) atomicAdd(lossacc, ls);
    }
}

// -----------------------------------------------------------------------------
// final: vq_loss = 1.25 * sum(diff^2)/16777216 ; perplexity from counts
// -----------------------------------------------------------------------------
__global__ void __launch_bounds__(256) vq_final(const unsigned* __restrict__ counts,
                                                const float* __restrict__ lossacc,
                                                float* __restrict__ out) {
    __shared__ float red[256];
    int t = threadIdx.x;
    float s = 0.f;
    for (int k = t; k < KCODE; k += 256) {
        float avg = (float)counts[k] * (1.0f / 65536.0f);
        s += avg * logf(fmaxf(avg, 1e-10f));
    }
    red[t] = s;
    __syncthreads();
    for (int st = 128; st >= 1; st >>= 1) {
        if (t < st) red[t] += red[t + st];
        __syncthreads();
    }
    if (t == 0) {
        out[LOSS_POS] = 1.25f * (*lossacc) * (1.0f / 16777216.0f);
        out[PERP_POS] = expf(-red[0]);
    }
}

// -----------------------------------------------------------------------------
extern "C" void kernel_launch(void* const* d_in, const int* in_sizes, int n_in,
                              void* d_out, int out_size, void* d_ws, size_t ws_size,
                              hipStream_t stream) {
    (void)in_sizes; (void)n_in; (void)out_size; (void)ws_size;
    const float* z   = (const float*)d_in[0];   // [16,256,64,64] f32
    const float* emb = (const float*)d_in[1];   // [1024,256] f32
    char* ws = (char*)d_ws;
    float*     lossacc = (float*)(ws + WS_LOSS);
    unsigned*  counts  = (unsigned*)(ws + WS_COUNTS);
    float*     enorm2  = (float*)(ws + WS_ENORM);
    _Float16*  ebB     = (_Float16*)(ws + WS_EBB);
    float* out = (float*)d_out;

    vq_prep <<<73, 256, 0, stream>>>(emb, ebB, enorm2, counts, lossacc);
    // LDS size is a host-side constant (independent of device-pass macros)
    vq_main <<<NBLK, TPB, LDS_BYTES, stream>>>(z, emb, ebB, enorm2,
                                               counts, lossacc, out);
    vq_final<<<1, 256, 0, stream>>>(counts, lossacc, out);
}
